// Recurrence_57372173140634
// MI455X (gfx1250) — matrix operand order repaired and checked by reading, not compiled, not run on hardware
//
#include <hip/hip_runtime.h>
#include <hip/hip_bf16.h>

// ---------------------------------------------------------------------------
// Problem constants (from the reference)
// ---------------------------------------------------------------------------
#define TT   128
#define NN   512
#define OBS  256
#define SS   64
#define HH   512
#define DD   321          // OBS + S + 1
#define G3   1536         // 3*H
#define HXW  517          // 5 + H
#define KIN  768          // OBS + H (MLP layer-1 K)

typedef __attribute__((ext_vector_type(16))) _Float16 v16h;
typedef __attribute__((ext_vector_type(8)))  _Float16 v8h;
typedef __attribute__((ext_vector_type(8)))  float    v8f;

// ---------------------------------------------------------------------------
// WMMA helpers (f16 inputs, f32 accumulate, 16x16x32)
// ---------------------------------------------------------------------------
__device__ __forceinline__ v8f wmma_f16(v16h a, v16h b, v8f c) {
    return __builtin_amdgcn_wmma_f32_16x16x32_f16(
        /*neg_a=*/false, a, /*neg_b=*/false, b,
        /*c_mod=*/(short)0, c, /*reuse_a=*/false, /*reuse_b=*/false);
}

// A fragment (16x32 f16) from row-major activations (ld = row stride, halfs).
// Layout: lanes 0-15: row M=lane, K={0..15}; lanes 16-31: K={16..31} (matches B).
__device__ __forceinline__ v16h load_a_frag(const _Float16* base, int ld) {
    int l = threadIdx.x & 31;
    const _Float16* p = base + (l & 15) * ld + ((l >> 4) << 4);
    v8h lo = *(const v8h*)(p);
    v8h hi = *(const v8h*)(p + 8);
    return __builtin_shufflevector(lo, hi, 0,1,2,3,4,5,6,7,8,9,10,11,12,13,14,15);
}

// B fragment (32x16 f16) for y = x @ W^T with W row-major (out x in, ld=in).
// Lane col needs 16 contiguous K halfs -> one 32B load.
__device__ __forceinline__ v16h load_b_frag(const _Float16* W, int ld, int col0, int kb) {
    int l = threadIdx.x & 31;
    int col = col0 + (l & 15);
    int k   = kb + ((l >> 4) << 4);
    return *(const v16h*)(W + (size_t)col * ld + k);
}

// ---------------------------------------------------------------------------
// Weight conversion f32 -> f16
// ---------------------------------------------------------------------------
__global__ void cvt_f16_kernel(const float* __restrict__ src,
                               _Float16* __restrict__ dst, int n) {
    int i = blockIdx.x * blockDim.x + threadIdx.x;
    if (i < n) dst[i] = (_Float16)src[i];
}

// ---------------------------------------------------------------------------
// GRU: 32 blocks x 16 batch rows, 16 waves (512 thr). h f16 in double-buffered
// LDS. Per step each wave owns 2 column slices; 3 gate accumulators per slice
// fused with sigmoid/tanh update in registers. Only final hT is written.
//
// NOTE: a zero offset is laundered through an empty asm each step so LICM
// cannot hoist the (step-invariant) Whh B-fragment loads out of the 64-step
// loop (hoisting spills ~3 KB/lane to scratch). Laundering an *offset* rather
// than the pointers keeps address-space inference intact, so the loads stay
// global_load_b128 (flat loads would tie up DScnt + the LDS issue path).
// ---------------------------------------------------------------------------
#define GRU_MR 16

__global__ __launch_bounds__(512) void gru_kernel(
    const float* __restrict__ inputs,
    const float* __restrict__ Wih,       // (3H,1) f32
    const float* __restrict__ bih,
    const float* __restrict__ bhh,
    const _Float16* __restrict__ Whh_h,  // (3H,H) f16 row-major
    _Float16* __restrict__ Mn_h)         // (N,H) f16 out
{
    __shared__ _Float16 hbuf[2][GRU_MR][HH];   // 32 KB
    __shared__ float    xs[SS][GRU_MR];        //  4 KB

    const int tid  = threadIdx.x;
    const int n0   = blockIdx.x * GRU_MR;
    const int wave = tid >> 5;                 // 0..15
    const int lane = tid & 31;
    const int colq = lane & 15;
    const int hi2  = lane >> 4;

    // stage x[s][i] = inputs[0, n0+i, OBS+s]
    for (int e = tid; e < SS * GRU_MR; e += 512) {
        int s = e / GRU_MR, i = e % GRU_MR;
        xs[s][i] = inputs[(size_t)(n0 + i) * DD + OBS + s];
    }
    _Float16* hz = &hbuf[0][0][0];
    for (int e = tid; e < GRU_MR * HH; e += 512) hz[e] = (_Float16)0.f;
    __syncthreads();

    size_t z = 0;                              // always 0, but opaque per step

    for (int s = 0; s < SS; ++s) {
        // defeat LICM while keeping global address-space inference
        asm volatile("" : "+s"(z));
        const _Float16* Whh_p = Whh_h + z;
        const float*    Wih_p = Wih + z;
        const float*    bih_p = bih + z;
        const float*    bhh_p = bhh + z;

        const _Float16* cur = &hbuf[s & 1][0][0];
        _Float16*       nxt = &hbuf[(s & 1) ^ 1][0][0];

        for (int q = 0; q < 2; ++q) {          // 16 waves x 2 = 32 slices
            const int c0 = (wave * 2 + q) * 16;
            v8f accr = {}, accz = {}, accn = {};
            for (int kb = 0; kb < HH; kb += 32) {
                v16h a  = load_a_frag(cur + kb, HH);
                v16h br = load_b_frag(Whh_p, HH, 0 * HH + c0, kb);
                v16h bz = load_b_frag(Whh_p, HH, 1 * HH + c0, kb);
                v16h bn = load_b_frag(Whh_p, HH, 2 * HH + c0, kb);
                accr = wmma_f16(a, br, accr);
                accz = wmma_f16(a, bz, accz);
                accn = wmma_f16(a, bn, accn);
            }
            const int cc = c0 + colq;
            const float wih_r = Wih_p[cc],       wih_z = Wih_p[HH + cc],
                        wih_n = Wih_p[2*HH + cc];
            const float bih_r = bih_p[cc],       bih_z = bih_p[HH + cc],
                        bih_n = bih_p[2*HH + cc];
            const float bhh_r = bhh_p[cc],       bhh_z = bhh_p[HH + cc],
                        bhh_n = bhh_p[2*HH + cc];
            v8f x8 = *(const v8f*)&xs[s][hi2 * 8];
            for (int v = 0; v < 8; ++v) {
                const int row = v + hi2 * 8;
                const float hold = (float)cur[row * HH + cc];
                const float ghr = accr[v] + bhh_r;
                const float ghz = accz[v] + bhh_z;
                const float ghn = accn[v] + bhh_n;
                const float ir  = x8[v] * wih_r + bih_r;
                const float iz  = x8[v] * wih_z + bih_z;
                const float inn = x8[v] * wih_n + bih_n;
                const float r = 1.f / (1.f + __expf(-(ir + ghr)));
                const float z2 = 1.f / (1.f + __expf(-(iz + ghz)));
                const float nv = tanhf(inn + r * ghn);
                nxt[row * HH + cc] = (_Float16)((1.f - z2) * nv + z2 * hold);
            }
        }
        __syncthreads();
    }
    const _Float16* fin = &hbuf[0][0][0];      // 64 steps -> buffer 0
    for (int e = tid; e < GRU_MR * HH; e += 512) {
        int i = e / HH, k = e % HH;
        Mn_h[(size_t)(n0 + i) * HH + k] = fin[e];
    }
}

// ---------------------------------------------------------------------------
// MLP heads: 1024 blocks x 64 rows (4 row-tiles), 8 waves. B fragments reused
// across 4 row-tiles (4x less weight traffic / issue). Scalar heads fused into
// layer-2 accumulators via shuffle reduction (no HC buffer).
// ---------------------------------------------------------------------------
#define TM 64
#define RT 4
#define MLP_LDS (TM*KIN*2 + TM*HH*2 + 8*TM*4)   // A0 + HB + partw = 165888 B

__device__ __forceinline__ void layer1_wmma(
    const _Float16* A0, int K,                 // LDS activations 64 x K
    const _Float16* __restrict__ W,            // f16 weights (512 x K)
    const float* __restrict__ bias,
    _Float16* HB, int wave, int lane)          // LDS out 64 x 512 (ReLU'd)
{
    const int colq = lane & 15, hi2 = lane >> 4;
    for (int q = 0; q < 4; ++q) {
        const int c0 = (wave * 4 + q) * 16;
        v8f acc[RT] = {{}, {}, {}, {}};
        for (int kb = 0; kb < K; kb += 32) {
            __builtin_prefetch(W + (size_t)(c0 + colq) * K + kb + 128, 0, 0);
            v16h b = load_b_frag(W, K, c0, kb);
#pragma unroll
            for (int rt = 0; rt < RT; ++rt) {
                v16h a = load_a_frag(A0 + (size_t)(rt * 16) * K + kb, K);
                acc[rt] = wmma_f16(a, b, acc[rt]);
            }
        }
        const int cc = c0 + colq;
        const float bv = bias[cc];
#pragma unroll
        for (int rt = 0; rt < RT; ++rt)
            for (int v = 0; v < 8; ++v)
                HB[(size_t)(rt * 16 + hi2 * 8 + v) * HH + cc] =
                    (_Float16)fmaxf(acc[rt][v] + bv, 0.f);
    }
}

__device__ __forceinline__ void layer2_head(
    const _Float16* HB,                        // LDS 64 x 512
    const _Float16* __restrict__ W,            // (512 x 512) f16
    const float* __restrict__ bias,
    const _Float16* __restrict__ hw, float hb0,// head vector + head bias
    float* __restrict__ outp, size_t r0,
    float* partw, int tid)
{
    const int wave = tid >> 5, lane = tid & 31;
    const int colq = lane & 15, hi2 = lane >> 4;
    float hp[RT][8];
#pragma unroll
    for (int rt = 0; rt < RT; ++rt)
        for (int v = 0; v < 8; ++v) hp[rt][v] = 0.f;

    for (int q = 0; q < 4; ++q) {
        const int c0 = (wave * 4 + q) * 16;
        v8f acc[RT] = {{}, {}, {}, {}};
        for (int kb = 0; kb < HH; kb += 32) {
            __builtin_prefetch(W + (size_t)(c0 + colq) * HH + kb + 128, 0, 0);
            v16h b = load_b_frag(W, HH, c0, kb);
#pragma unroll
            for (int rt = 0; rt < RT; ++rt) {
                v16h a = load_a_frag(HB + (size_t)(rt * 16) * HH + kb, HH);
                acc[rt] = wmma_f16(a, b, acc[rt]);
            }
        }
        const int cc = c0 + colq;
        const float bv = bias[cc];
        const float wv = (float)hw[cc];
#pragma unroll
        for (int rt = 0; rt < RT; ++rt)
            for (int v = 0; v < 8; ++v)
                hp[rt][v] += fmaxf(acc[rt][v] + bv, 0.f) * wv;
    }
    // reduce over the 16 columns (lane bits 0..3); hi2 halves stay separate
#pragma unroll
    for (int m = 1; m <= 8; m <<= 1)
#pragma unroll
        for (int rt = 0; rt < RT; ++rt)
            for (int v = 0; v < 8; ++v)
                hp[rt][v] += __shfl_xor(hp[rt][v], m, 32);
    if (colq == 0) {
#pragma unroll
        for (int rt = 0; rt < RT; ++rt)
            for (int v = 0; v < 8; ++v)
                partw[wave * TM + rt * 16 + hi2 * 8 + v] = hp[rt][v];
    }
    __syncthreads();
    if (tid < TM) {
        float s = hb0;
        for (int w = 0; w < 8; ++w) s += partw[w * TM + tid];
        outp[r0 + tid] = s;
    }
    __syncthreads();
}

__global__ __launch_bounds__(256) void mlp_kernel(
    const float* __restrict__ inputs,
    const _Float16* __restrict__ Mn_h,
    const _Float16* __restrict__ W1a, const float* __restrict__ b1a,
    const _Float16* __restrict__ W2a, const float* __restrict__ b2a,
    const _Float16* __restrict__ W1c, const float* __restrict__ b1c,
    const _Float16* __restrict__ W2c, const float* __restrict__ b2c,
    const _Float16* __restrict__ mW,  const float* __restrict__ mb,
    const _Float16* __restrict__ oW,  const float* __restrict__ ob,
    float* __restrict__ loc_buf, float* __restrict__ v_buf)
{
    extern __shared__ char dsm[];
    _Float16* A0    = (_Float16*)dsm;                            // 64 x 768
    _Float16* HB    = (_Float16*)(dsm + (size_t)TM * KIN * 2);   // 64 x 512
    float*    partw = (float*)(dsm + (size_t)TM * KIN * 2
                                   + (size_t)TM * HH * 2);       // 8 x 64

    const int tid  = threadIdx.x;
    const int wave = tid >> 5;
    const int lane = tid & 31;
    const size_t r0 = (size_t)blockIdx.x * TM;

    // stage A0 = [Mn(n)(512) || f16(obs[t,n])(256)] per row
    for (int e = tid; e < TM * KIN; e += 256) {
        int i = e / KIN, c = e % KIN;
        size_t r = r0 + i;
        int t = (int)(r >> 9), n = (int)(r & (NN - 1));
        A0[e] = (c < HH) ? Mn_h[(size_t)n * HH + c]
                         : (_Float16)inputs[((size_t)t * NN + n) * DD + (c - HH)];
    }
    __syncthreads();

    // actor
    layer1_wmma(A0, KIN, W1a, b1a, HB, wave, lane);
    __syncthreads();
    layer2_head(HB, W2a, b2a, mW, mb[0], loc_buf, r0, partw, tid);
    // critic
    layer1_wmma(A0, KIN, W1c, b1c, HB, wave, lane);
    __syncthreads();
    layer2_head(HB, W2c, b2c, oW, ob[0], v_buf, r0, partw, tid);
}

// ---------------------------------------------------------------------------
// Output assembly: hx = [a, loc, scale, v, h_hidden(512), p_out] per (t,n),
// plus duplicated hx[-1:] slice (rows 128*512 .. 129*512-1 use t=127).
// ---------------------------------------------------------------------------
__global__ __launch_bounds__(256) void assemble_kernel(
    const float* __restrict__ inputs,
    const float* __restrict__ rnn_hxs,
    const float* __restrict__ logstd,
    const float* __restrict__ loc_buf,
    const float* __restrict__ v_buf,
    float* __restrict__ out)
{
    const int R  = blockIdx.x;          // 0 .. 129*512-1
    const int tt = R >> 9;
    const int n  = R & (NN - 1);
    const int t  = (tt == TT) ? (TT - 1) : tt;
    float* o = out + (size_t)R * HXW;
    const float* hx = rnn_hxs + (size_t)n * HXW;
    const size_t row = (size_t)t * NN + n;

    if (threadIdx.x == 0) {
        o[0] = inputs[row * DD + (DD - 1)];              // action
        o[1] = loc_buf[row];                              // loc
        o[2] = __expf(logstd[0]);                         // scale
        o[3] = v_buf[row];                                // value
        int P = (int)hx[HXW - 1];
        o[HXW - 1] = (float)((P + 1) % SS);               // p_out
    }
    for (int k = threadIdx.x; k < HH; k += blockDim.x)
        o[4 + k] = hx[4 + k];                             // h_hidden broadcast
}

// ---------------------------------------------------------------------------
// Launch
// ---------------------------------------------------------------------------
extern "C" void kernel_launch(void* const* d_in, const int* in_sizes, int n_in,
                              void* d_out, int out_size, void* d_ws, size_t ws_size,
                              hipStream_t stream) {
    const float* inputs    = (const float*)d_in[0];
    const float* rnn_hxs   = (const float*)d_in[1];
    const float* gru_Wih   = (const float*)d_in[2];
    const float* gru_Whh   = (const float*)d_in[3];
    const float* gru_bih   = (const float*)d_in[4];
    const float* gru_bhh   = (const float*)d_in[5];
    const float* actor_W1  = (const float*)d_in[6];
    const float* actor_b1  = (const float*)d_in[7];
    const float* actor_W2  = (const float*)d_in[8];
    const float* actor_b2  = (const float*)d_in[9];
    const float* critic_W1 = (const float*)d_in[10];
    const float* critic_b1 = (const float*)d_in[11];
    const float* critic_W2 = (const float*)d_in[12];
    const float* critic_b2 = (const float*)d_in[13];
    const float* mean_W    = (const float*)d_in[14];
    const float* mean_b    = (const float*)d_in[15];
    const float* logstd    = (const float*)d_in[16];
    const float* out_W     = (const float*)d_in[17];
    const float* out_b     = (const float*)d_in[18];
    float* out = (float*)d_out;

    // workspace carve-up (f16 weights + Mn, then f32 head outputs)
    _Float16* p16   = (_Float16*)d_ws;
    _Float16* Whh_h = p16;  p16 += G3 * HH;
    _Float16* W1a_h = p16;  p16 += HH * KIN;
    _Float16* W2a_h = p16;  p16 += HH * HH;
    _Float16* W1c_h = p16;  p16 += HH * KIN;
    _Float16* W2c_h = p16;  p16 += HH * HH;
    _Float16* mW_h  = p16;  p16 += HH;
    _Float16* oW_h  = p16;  p16 += HH;
    _Float16* Mn_h  = p16;  p16 += (size_t)NN * HH;
    float* loc_buf  = (float*)p16;
    float* v_buf    = loc_buf + (size_t)TT * NN;

    const int TH = 256;
    auto cvt = [&](const float* s, _Float16* d, int n) {
        cvt_f16_kernel<<<(n + TH - 1) / TH, TH, 0, stream>>>(s, d, n);
    };
    cvt(gru_Whh,   Whh_h, G3 * HH);
    cvt(actor_W1,  W1a_h, HH * KIN);
    cvt(actor_W2,  W2a_h, HH * HH);
    cvt(critic_W1, W1c_h, HH * KIN);
    cvt(critic_W2, W2c_h, HH * HH);
    cvt(mean_W,    mW_h,  HH);
    cvt(out_W,     oW_h,  HH);

    gru_kernel<<<NN / GRU_MR, 512, 0, stream>>>(
        inputs, gru_Wih, gru_bih, gru_bhh, Whh_h, Mn_h);

    mlp_kernel<<<(TT * NN) / TM, TH, MLP_LDS, stream>>>(
        inputs, Mn_h,
        W1a_h, actor_b1, W2a_h, actor_b2,
        W1c_h, critic_b1, W2c_h, critic_b2,
        mW_h, mean_b, oW_h, out_b,
        loc_buf, v_buf);

    assemble_kernel<<<(TT + 1) * NN, TH, 0, stream>>>(
        inputs, rnn_hxs, logstd, loc_buf, v_buf, out);
}
